// PriorFusion2D_17119739642223
// MI455X (gfx1250) — matrix operand order repaired and checked by reading, not compiled, not run on hardware
//
#include <hip/hip_runtime.h>

// ---------------------------------------------------------------------------
// PriorFusion2D for MI455X (gfx1250, wave32, WMMA bf16 16x16x32, f32 accum)
// Register-blocked implicit-GEMM: 32x64 output tile per wave, 8 accumulators.
// ---------------------------------------------------------------------------

typedef __bf16 v16bf __attribute__((ext_vector_type(16)));
typedef float  v8f   __attribute__((ext_vector_type(8)));

#define XD   100
#define YD   200
#define ZD   16
#define ZPD  4
#define HIDN 256
#define BCN  256
#define BSN  2
#define NPTS 40000
#define VCN  68
#define VCP  96      // 68 padded to multiple of 32
#define BN_EPS 1e-5f

static __device__ __forceinline__ unsigned short f2bf(float f) {
    union { float f; unsigned u; } c; c.f = f;
    unsigned u = c.u;
    unsigned r = u + 0x7FFFu + ((u >> 16) & 1u);   // round-to-nearest-even
    return (unsigned short)(r >> 16);
}
static __device__ __forceinline__ float bf2f(unsigned short h) {
    union { unsigned u; float f; } c; c.u = ((unsigned)h) << 16;
    return c.f;
}

// ---------------- elementwise conversion / packing kernels -----------------

__global__ void k_f32_to_bf16_pad(const float* __restrict__ src,
                                  unsigned short* __restrict__ dst,
                                  int M, int K, int Kp) {
    long long t = (long long)blockIdx.x * blockDim.x + threadIdx.x;
    long long n = (long long)M * Kp;
    if (t >= n) return;
    int k = (int)(t % Kp);
    long long m = t / Kp;
    dst[t] = (k < K) ? f2bf(src[m * K + k]) : (unsigned short)0;
}

// (K x N) f32 -> (N x Kp) bf16 (transposed, zero padded K)
__global__ void k_pack_linear(const float* __restrict__ src,
                              unsigned short* __restrict__ dst,
                              int K, int N, int Kp) {
    long long t = (long long)blockIdx.x * blockDim.x + threadIdx.x;
    long long n = (long long)N * Kp;
    if (t >= n) return;
    int k = (int)(t % Kp);
    int nn = (int)(t / Kp);
    dst[t] = (k < K) ? f2bf(src[(long long)k * N + nn]) : (unsigned short)0;
}

// (O, I, KH, KW) f32 -> (KH*KW, O, I) bf16
__global__ void k_pack_conv(const float* __restrict__ src,
                            unsigned short* __restrict__ dst,
                            int O, int I, int KH, int KW) {
    long long t = (long long)blockIdx.x * blockDim.x + threadIdx.x;
    long long n = (long long)KH * KW * O * I;
    if (t >= n) return;
    int i  = (int)(t % I);
    long long r = t / I;
    int o  = (int)(r % O);
    int tp = (int)(r / O);
    int kh = tp / KW, kw = tp % KW;
    dst[t] = f2bf(src[(((long long)o * I + i) * KH + kh) * KW + kw]);
}

__global__ void k_fill_i32(int* __restrict__ p, long long n, int v) {
    long long t = (long long)blockIdx.x * blockDim.x + threadIdx.x;
    if (t < n) p[t] = v;
}

// last-write-wins == max point index wins
__global__ void k_scatter_idx(const int* __restrict__ coords, int* __restrict__ idxg) {
    long long t = (long long)blockIdx.x * blockDim.x + threadIdx.x;
    if (t >= (long long)BSN * NPTS) return;
    int b = (int)(t / NPTS);
    int i = (int)(t % NPTS);
    int cx = coords[t * 3 + 0];
    int cy = coords[t * 3 + 1];
    int cz = coords[t * 3 + 2];
    atomicMax(&idxg[(((long long)b * XD + cx) * YD + cy) * ZD + cz], i);
}

// Fused gather + z-maxpool: pooled (bs, Y, X, HID*ZP) bf16, channel = hid*4+zp
__global__ void k_pool_build(const int* __restrict__ idxg,
                             const unsigned short* __restrict__ h2,
                             unsigned short* __restrict__ pooled) {
    const int p = blockIdx.x;                 // bs*Y*X blocks
    const int x = p % XD;
    const int y = (p / XD) % YD;
    const int b = p / (XD * YD);
    __shared__ int sidx[ZD];
    if (threadIdx.x < ZD)
        sidx[threadIdx.x] = idxg[(((long long)b * XD + x) * YD + y) * ZD + threadIdx.x];
    __syncthreads();
    const int hid = threadIdx.x;              // 256 threads
    unsigned short outv[ZPD];
    #pragma unroll
    for (int zp = 0; zp < ZPD; ++zp) {
        unsigned short best = 0;              // bf16 0.0 (empty cells; values >= 0)
        float bestf = 0.0f;
        #pragma unroll
        for (int dz = 0; dz < 4; ++dz) {
            int w = sidx[zp * 4 + dz];
            if (w >= 0) {
                unsigned short us = h2[((long long)b * NPTS + w) * HIDN + hid];
                float f = bf2f(us);
                if (f > bestf) { bestf = f; best = us; }
            }
        }
        outv[zp] = best;
    }
    unsigned short* dst = pooled + (((long long)b * YD + y) * XD + x) * (HIDN * ZPD) + hid * ZPD;
    #pragma unroll
    for (int zp = 0; zp < ZPD; ++zp) dst[zp] = outv[zp];
}

// (bs, C, H, W) f32 -> (bs, H, W, C) bf16
__global__ void k_nchw_to_nhwc(const float* __restrict__ src,
                               unsigned short* __restrict__ dst,
                               int C, int H, int W) {
    long long t = (long long)blockIdx.x * blockDim.x + threadIdx.x;
    long long n = (long long)BSN * C * H * W;
    if (t >= n) return;
    int c = (int)(t % C);
    long long r = t / C;
    int x = (int)(r % W); r /= W;
    int y = (int)(r % H);
    int b = (int)(r / H);
    dst[t] = f2bf(src[(((long long)b * C + c) * H + y) * W + x]);
}

// ------------------------- WMMA implicit-GEMM conv -------------------------
// D = concat(inA, inB) (*) Wp + bias, over KHxKW taps (same-pad).
// inA/inB: (bs, H, W, C*) bf16 NHWC.  Wp: (KH*KW, N, CA+CB) bf16 (weight^T).
// Each wave computes a 32(M) x 64(N) tile: 2x4 sub-tiles of 16x16, K consumed
// 32 at a time -> 8 independent v_wmma_f32_16x16x32_bf16 per K-step
// (~22 FLOP/byte of fragment traffic, independent accumulator chains).
// Requires M%32==0 (batch boundary 32-aligned), N%64==0, C%32==0.
__global__ void k_conv_wmma(const unsigned short* __restrict__ inA, int CA,
                            const unsigned short* __restrict__ inB, int CB,
                            const unsigned short* __restrict__ Wp, int ldw,
                            const float* __restrict__ bias,
                            float* __restrict__ outF,
                            unsigned short* __restrict__ outB,
                            int M, int N, int H, int Wd, int KH, int KW,
                            int relu) {
    const int lane = threadIdx.x & 31;
    const int wave = blockIdx.x * (blockDim.x >> 5) + (threadIdx.x >> 5);
    const int tilesN = N >> 6;                 // 64-wide N tiles
    const int tilesM = M >> 5;                 // 32-tall M tiles
    if (wave >= tilesM * tilesN) return;
    const int tm = wave / tilesN, tn = wave % tilesN;
    const int m0 = tm << 5, n0 = tn << 6;
    const int li = lane & 15, hi = lane >> 4;
    const int HW = H * Wd;
    const int PH = KH >> 1, PW = KW >> 1;

    // Two A-row owners per lane (sub-tiles i=0,1): rows m0+li, m0+16+li
    int bb[2], yy0[2], xx0[2];
    #pragma unroll
    for (int i = 0; i < 2; ++i) {
        const int mrow = m0 + (i << 4) + li;
        bb[i]  = mrow / HW;
        const int rem = mrow % HW;
        yy0[i] = rem / Wd;
        xx0[i] = rem % Wd;
    }

    // ISA 16-bit A 16x32 per-lane k map (pairs of bf16 -> dword loads)
    int kA[8];
    #pragma unroll
    for (int p = 0; p < 8; ++p)
        kA[p] = ((p < 4) ? 0 : 16) + hi * 8 + ((p & 3) << 1);

    v8f acc[2][4];
    #pragma unroll
    for (int i = 0; i < 2; ++i)
        #pragma unroll
        for (int j = 0; j < 4; ++j)
            acc[i][j] = (v8f){0.f, 0.f, 0.f, 0.f, 0.f, 0.f, 0.f, 0.f};

    const int T = KH * KW;
    for (int t = 0; t < T; ++t) {
        const int kh = t / KW, kw = t % KW;
        // per-subtile tap validity + row pointers
        bool valid[2];
        const unsigned short* arowbase[2];
        #pragma unroll
        for (int i = 0; i < 2; ++i) {
            const int yy = yy0[i] + kh - PH, xx = xx0[i] + kw - PW;
            valid[i] = (yy >= 0) && (yy < H) && (xx >= 0) && (xx < Wd);
            arowbase[i] = (const unsigned short*)((long long)((long long)(bb[i] * H + yy) * Wd + xx));
        }
        // weight^T row pointers for the 4 N sub-tiles
        const unsigned short* wrow[4];
        #pragma unroll
        for (int j = 0; j < 4; ++j)
            wrow[j] = Wp + (long long)(t * N + n0 + (j << 4) + li) * ldw;

        for (int seg = 0; seg < 2; ++seg) {
            const unsigned short* src = (seg == 0) ? inA : inB;
            const int Cs  = (seg == 0) ? CA : CB;
            const int kwb = (seg == 0) ? 0 : CA;
            if (src == nullptr || Cs <= 0) continue;
            const unsigned short* arow[2];
            #pragma unroll
            for (int i = 0; i < 2; ++i)
                arow[i] = src + (long long)arowbase[i] * Cs;

            for (int k0 = 0; k0 < Cs; k0 += 32) {
                union { v16bf v; unsigned u[8]; } fa[2], fb[4];
                #pragma unroll
                for (int i = 0; i < 2; ++i) {
                    const unsigned short* pa = arow[i] + k0;
                    #pragma unroll
                    for (int p = 0; p < 8; ++p)
                        fa[i].u[p] = valid[i] ? *(const unsigned*)(pa + kA[p]) : 0u;
                }
                #pragma unroll
                for (int j = 0; j < 4; ++j) {
                    // B 32x16: lane = column n, k = hi*8 + {0..7} and 16 + hi*8 + {0..7} (matches A)
                    const unsigned short* pb = wrow[j] + kwb + k0 + hi * 8;
                    __builtin_prefetch(pb + 32, 0, 1);
                    #pragma unroll
                    for (int p = 0; p < 8; ++p)
                        fb[j].u[p] = *(const unsigned*)(pb + ((p < 4) ? 0 : 16) + ((p & 3) << 1));
                }
                #pragma unroll
                for (int j = 0; j < 4; ++j)
                    #pragma unroll
                    for (int i = 0; i < 2; ++i)
                        acc[i][j] = __builtin_amdgcn_wmma_f32_16x16x32_bf16(
                                        false, fa[i].v, false, fb[j].v,
                                        (short)0, acc[i][j], false, false);
            }
        }
    }

    // C/D layout: lane = column (sub)n0+li, VGPR r -> row (sub)m0 + r + 8*hi
    #pragma unroll
    for (int j = 0; j < 4; ++j) {
        const int nn = n0 + (j << 4) + li;
        const float bz = bias ? bias[nn] : 0.0f;
        #pragma unroll
        for (int i = 0; i < 2; ++i) {
            #pragma unroll
            for (int r = 0; r < 8; ++r) {
                float v = acc[i][j][r] + bz;
                if (relu) v = fmaxf(v, 0.0f);
                const long long mm = m0 + (i << 4) + hi * 8 + r;
                const long long o  = mm * N + nn;
                if (outF) outF[o] = v;
                if (outB) outB[o] = f2bf(v);
            }
        }
    }
}

// ------------------------- BatchNorm (training mode) -----------------------
// Stage 1: partial sum / sumsq per channel per row-split (deterministic tree).
__global__ void k_bn_stats(const float* __restrict__ x,
                           float* __restrict__ part, int M, int N) {
    const int c  = blockIdx.x * 64 + (threadIdx.x & 63);
    const int rg = threadIdx.x >> 6;          // 0..3
    const int SPLIT = gridDim.y;
    float s = 0.f, ss = 0.f;
    for (int m = blockIdx.y * 4 + rg; m < M; m += SPLIT * 4) {
        float v = x[(long long)m * N + c];
        s += v; ss += v * v;
    }
    __shared__ float sh0[4][64];
    __shared__ float sh1[4][64];
    sh0[rg][threadIdx.x & 63] = s;
    sh1[rg][threadIdx.x & 63] = ss;
    __syncthreads();
    if (rg == 0) {
        int l = threadIdx.x;                  // 0..63
        float fs  = sh0[0][l] + sh0[1][l] + sh0[2][l] + sh0[3][l];
        float fss = sh1[0][l] + sh1[1][l] + sh1[2][l] + sh1[3][l];
        part[(long long)blockIdx.y * (2 * N) + c]     = fs;
        part[(long long)blockIdx.y * (2 * N) + N + c] = fss;
    }
}

__global__ void k_bn_reduce(const float* __restrict__ part,
                            float* __restrict__ stats, int n2, int splits) {
    int t = blockIdx.x * blockDim.x + threadIdx.x;
    if (t >= n2) return;
    float s = 0.f;
    for (int i = 0; i < splits; ++i) s += part[(long long)i * n2 + t];
    stats[t] = s;
}

// BN + ReLU -> bf16 NHWC (feeds next WMMA conv)
__global__ void k_bn_apply_bf16(const float* __restrict__ x,
                                const float* __restrict__ stats,
                                const float* __restrict__ g,
                                const float* __restrict__ be,
                                unsigned short* __restrict__ out,
                                int M, int N) {
    long long t = (long long)blockIdx.x * blockDim.x + threadIdx.x;
    long long n = (long long)M * N;
    if (t >= n) return;
    int c = (int)(t % N);
    float mean = stats[c] / (float)M;
    float var  = stats[N + c] / (float)M - mean * mean;   // biased, as jnp.var
    float sc = g[c] * rsqrtf(var + BN_EPS);
    float v  = (x[t] - mean) * sc + be[c];
    out[t] = f2bf(fmaxf(v, 0.0f));
}

// Final BN + ReLU, NHWC f32 -> NCHW f32 (reference output layout)
__global__ void k_bn_final(const float* __restrict__ x,
                           const float* __restrict__ stats,
                           const float* __restrict__ g,
                           const float* __restrict__ be,
                           float* __restrict__ out, int C, int H, int W) {
    long long t = (long long)blockIdx.x * blockDim.x + threadIdx.x;
    long long n = (long long)BSN * C * H * W;
    if (t >= n) return;
    int xw = (int)(t % W);
    long long r = t / W;
    int y = (int)(r % H); r /= H;
    int c = (int)(r % C);
    int b = (int)(r / C);
    const int M = BSN * H * W;
    float mean = stats[c] / (float)M;
    float var  = stats[C + c] / (float)M - mean * mean;
    float sc = g[c] * rsqrtf(var + BN_EPS);
    float v  = x[(((long long)b * H + y) * W + xw) * C + c];
    v = (v - mean) * sc + be[c];
    out[t] = fmaxf(v, 0.0f);
}

// ------------------------------- launcher ----------------------------------

static inline unsigned gblk(long long n) { return (unsigned)((n + 255) / 256); }

extern "C" void kernel_launch(void* const* d_in, const int* in_sizes, int n_in,
                              void* d_out, int out_size, void* d_ws, size_t ws_size,
                              hipStream_t stream) {
    (void)in_sizes; (void)n_in; (void)out_size; (void)ws_size;
    const float* bev = (const float*)d_in[0];
    const float* pf  = (const float*)d_in[1];
    const float* w1  = (const float*)d_in[2];
    const float* b1  = (const float*)d_in[3];
    const float* w2  = (const float*)d_in[4];
    const float* b2  = (const float*)d_in[5];
    const float* c1w = (const float*)d_in[6];
    const float* c1b = (const float*)d_in[7];
    const float* c2w = (const float*)d_in[8];
    const float* c2b = (const float*)d_in[9];
    const float* c3w = (const float*)d_in[10];
    const float* c3b = (const float*)d_in[11];
    const float* c4w = (const float*)d_in[12];
    const float* c4b = (const float*)d_in[13];
    const float* g1  = (const float*)d_in[14]; const float* be1 = (const float*)d_in[15];
    const float* g2  = (const float*)d_in[16]; const float* be2 = (const float*)d_in[17];
    const float* g3  = (const float*)d_in[18]; const float* be3 = (const float*)d_in[19];
    const float* g4  = (const float*)d_in[20]; const float* be4 = (const float*)d_in[21];
    const int* coords = (const int*)d_in[22];
    float* out = (float*)d_out;

    char* base = (char*)d_ws;
    size_t off = 0;
    auto wsA = [&](size_t bytes) -> char* {
        char* p = base + off;
        off += (bytes + 255) & ~(size_t)255;
        return p;
    };

    const long long Mml = (long long)BSN * NPTS;        // 80000
    const long long Mcv = (long long)BSN * YD * XD;     // 40000

    unsigned short* Aml    = (unsigned short*)wsA(Mml * VCP * 2);
    unsigned short* w1p    = (unsigned short*)wsA((size_t)HIDN * VCP * 2);
    unsigned short* w2p    = (unsigned short*)wsA((size_t)HIDN * HIDN * 2);
    unsigned short* c1p    = (unsigned short*)wsA((size_t)BCN * (HIDN * ZPD) * 2);
    unsigned short* c2p    = (unsigned short*)wsA((size_t)9 * HIDN * HIDN * 2);
    unsigned short* c3p    = (unsigned short*)wsA((size_t)BCN * (HIDN + BCN) * 2);
    unsigned short* c4p    = (unsigned short*)wsA((size_t)9 * BCN * BCN * 2);
    unsigned short* h1     = (unsigned short*)wsA(Mml * HIDN * 2);
    unsigned short* h2     = (unsigned short*)wsA(Mml * HIDN * 2);
    int*            idxg   = (int*)wsA((size_t)BSN * XD * YD * ZD * 4);
    unsigned short* pooled = (unsigned short*)wsA(Mcv * (HIDN * ZPD) * 2);
    unsigned short* bevb   = (unsigned short*)wsA(Mcv * BCN * 2);
    float*          convO  = (float*)wsA(Mcv * BCN * 4);
    unsigned short* actA   = (unsigned short*)wsA(Mcv * BCN * 2);
    unsigned short* actB   = (unsigned short*)wsA(Mcv * BCN * 2);
    float*          part   = (float*)wsA((size_t)40 * 512 * 4);
    float*          stats  = (float*)wsA((size_t)512 * 4);

    // --- pack weights / convert inputs to bf16 ---
    k_f32_to_bf16_pad<<<gblk(Mml * VCP), 256, 0, stream>>>(pf, Aml, (int)Mml, VCN, VCP);
    k_pack_linear<<<gblk((long long)HIDN * VCP), 256, 0, stream>>>(w1, w1p, VCN, HIDN, VCP);
    k_pack_linear<<<gblk((long long)HIDN * HIDN), 256, 0, stream>>>(w2, w2p, HIDN, HIDN, HIDN);
    k_pack_conv<<<gblk((long long)BCN * HIDN * ZPD), 256, 0, stream>>>(c1w, c1p, BCN, HIDN * ZPD, 1, 1);
    k_pack_conv<<<gblk((long long)9 * HIDN * HIDN), 256, 0, stream>>>(c2w, c2p, HIDN, HIDN, 3, 3);
    k_pack_conv<<<gblk((long long)BCN * (HIDN + BCN)), 256, 0, stream>>>(c3w, c3p, BCN, HIDN + BCN, 1, 1);
    k_pack_conv<<<gblk((long long)9 * BCN * BCN), 256, 0, stream>>>(c4w, c4p, BCN, BCN, 3, 3);

    // --- point MLP (two WMMA GEMMs, relu, bf16 out) ---
    const int wavesMl = (int)((Mml / 32) * (HIDN / 64));     // 2500*4 = 10000
    const unsigned gMl = (unsigned)((wavesMl + 7) / 8);
    k_conv_wmma<<<gMl, 256, 0, stream>>>(Aml, VCP, nullptr, 0, w1p, VCP, b1,
                                         nullptr, h1, (int)Mml, HIDN, (int)Mml, 1, 1, 1, 1);
    k_conv_wmma<<<gMl, 256, 0, stream>>>(h1, HIDN, nullptr, 0, w2p, HIDN, b2,
                                         nullptr, h2, (int)Mml, HIDN, (int)Mml, 1, 1, 1, 1);

    // --- scatter winners + fused z-maxpool into pooled NHWC bf16 ---
    const long long nIdx = (long long)BSN * XD * YD * ZD;
    k_fill_i32<<<gblk(nIdx), 256, 0, stream>>>(idxg, nIdx, -1);
    k_scatter_idx<<<gblk(Mml), 256, 0, stream>>>(coords, idxg);
    k_pool_build<<<(unsigned)(BSN * YD * XD), 256, 0, stream>>>(idxg, h2, pooled);
    k_nchw_to_nhwc<<<gblk(Mcv * BCN), 256, 0, stream>>>(bev, bevb, BCN, YD, XD);

    const int wavesCv = (int)((Mcv / 32) * (BCN / 64));      // 1250*4 = 5000
    const unsigned gCv = (unsigned)((wavesCv + 7) / 8);
    const dim3 sg(BCN / 64, 40);
    const unsigned gEw = gblk(Mcv * BCN);

    // --- block1: conv1 (1x1, 1024->256) + BN + ReLU ---
    k_conv_wmma<<<gCv, 256, 0, stream>>>(pooled, HIDN * ZPD, nullptr, 0, c1p, HIDN * ZPD,
                                         c1b, convO, nullptr, (int)Mcv, BCN, YD, XD, 1, 1, 0);
    k_bn_stats<<<sg, 256, 0, stream>>>(convO, part, (int)Mcv, BCN);
    k_bn_reduce<<<2, 256, 0, stream>>>(part, stats, 2 * BCN, 40);
    k_bn_apply_bf16<<<gEw, 256, 0, stream>>>(convO, stats, g1, be1, actA, (int)Mcv, BCN);

    // --- block1: conv2 (3x3, 256->256) + BN + ReLU ---
    k_conv_wmma<<<gCv, 256, 0, stream>>>(actA, HIDN, nullptr, 0, c2p, HIDN,
                                         c2b, convO, nullptr, (int)Mcv, BCN, YD, XD, 3, 3, 0);
    k_bn_stats<<<sg, 256, 0, stream>>>(convO, part, (int)Mcv, BCN);
    k_bn_reduce<<<2, 256, 0, stream>>>(part, stats, 2 * BCN, 40);
    k_bn_apply_bf16<<<gEw, 256, 0, stream>>>(convO, stats, g2, be2, actB, (int)Mcv, BCN);

    // --- block2: conv3 (1x1 on concat(bev, p), 512->256) + BN + ReLU ---
    k_conv_wmma<<<gCv, 256, 0, stream>>>(bevb, BCN, actB, HIDN, c3p, HIDN + BCN,
                                         c3b, convO, nullptr, (int)Mcv, BCN, YD, XD, 1, 1, 0);
    k_bn_stats<<<sg, 256, 0, stream>>>(convO, part, (int)Mcv, BCN);
    k_bn_reduce<<<2, 256, 0, stream>>>(part, stats, 2 * BCN, 40);
    k_bn_apply_bf16<<<gEw, 256, 0, stream>>>(convO, stats, g3, be3, actA, (int)Mcv, BCN);

    // --- block2: conv4 (3x3, 256->256) + BN, final ReLU -> NCHW f32 ---
    k_conv_wmma<<<gCv, 256, 0, stream>>>(actA, BCN, nullptr, 0, c4p, BCN,
                                         c4b, convO, nullptr, (int)Mcv, BCN, YD, XD, 3, 3, 0);
    k_bn_stats<<<sg, 256, 0, stream>>>(convO, part, (int)Mcv, BCN);
    k_bn_reduce<<<2, 256, 0, stream>>>(part, stats, 2 * BCN, 40);
    k_bn_final<<<gblk(Mcv * BCN), 256, 0, stream>>>(convO, stats, g4, be4, out, BCN, YD, XD);
}
